// GraphEncoder_84774064488559
// MI455X (gfx1250) — compile-verified, matrix operand order checked by reading, not run on hardware
//
#include <hip/hip_runtime.h>
#include <hip/hip_bf16.h>
#include <math.h>

#define DM 768
#define BK 32
#define BM 128
#define BN 128
#define LDSP 40   // 80-byte rows: 16B-aligned fragment chunks, 20-dword stride (conflict-free)

typedef __attribute__((ext_vector_type(16))) __bf16 v16bf;
typedef __attribute__((ext_vector_type(8)))  __bf16 v8bf;
typedef __attribute__((ext_vector_type(4)))  __bf16 v4bf;
typedef __attribute__((ext_vector_type(8)))  float  v8f;
typedef __attribute__((ext_vector_type(4)))  unsigned int v4ui;

static __device__ __forceinline__ float gelu_exact(float x) {
    return 0.5f * x * (1.0f + erff(x * 0.70710678118654752f));
}

union Frag { v16bf v; v4ui q[2]; };

// C[M,768] = act( (A (+A2)) @ W[768,768] + bias ), bf16 WMMA, fp32 accum
// ACT: 0 = none, 1 = relu, 2 = exact gelu
// Software pipeline: global loads for tile k+1 issued before tile k's WMMAs.
template<int ACT, bool HAS_A2>
__global__ __launch_bounds__(256) void gemm_wmma(
    const float* __restrict__ A, const float* __restrict__ A2,
    const float* __restrict__ W, const float* __restrict__ bias,
    float* __restrict__ C, int M)
{
    __shared__ __bf16 As[BM][LDSP];   // [m][k]
    __shared__ __bf16 Bs[BN][LDSP];   // [n][k] (W transposed)

    const int t    = threadIdx.x;
    const int lane = t & 31;
    const int wave = t >> 5;
    const int half = lane >> 4;
    const int l16  = lane & 15;
    const int mBase = blockIdx.x * BM;
    const int nBase = blockIdx.y * BN;

    v8f acc[8] = {};

    // A staging map: 4x float4 per thread, row = t/8 (+32*i), col4 = (t%8)*4
    const int ar0 = t >> 3;
    const int ac0 = (t & 7) << 2;
    // B staging map: thread = one column n = t%128, k-range = (t/128)*16 .. +15
    const int bn  = t & 127;
    const int bk0 = (t >> 7) << 4;

    float4 aReg[4];
    float  bReg[16];

    auto globalLoad = [&](int kBase) {
        #pragma unroll
        for (int i = 0; i < 4; ++i) {
            int row = mBase + ar0 + i * 32;
            float4 v = {0.f, 0.f, 0.f, 0.f};
            if (row < M) {
                v = *(const float4*)(A + (size_t)row * DM + kBase + ac0);
                if (HAS_A2) {
                    float4 v2 = *(const float4*)(A2 + (size_t)row * DM + kBase + ac0);
                    v.x += v2.x; v.y += v2.y; v.z += v2.z; v.w += v2.w;
                }
            }
            aReg[i] = v;
        }
        // column-major W reads: coalesced across 128 consecutive-n lanes per k row
        #pragma unroll
        for (int kk = 0; kk < 16; ++kk)
            bReg[kk] = W[(size_t)(kBase + bk0 + kk) * DM + nBase + bn];
    };

    auto ldsStore = [&]() {
        #pragma unroll
        for (int i = 0; i < 4; ++i) {
            int r = ar0 + i * 32;
            v4bf p;
            p.x = (__bf16)aReg[i].x; p.y = (__bf16)aReg[i].y;
            p.z = (__bf16)aReg[i].z; p.w = (__bf16)aReg[i].w;
            *(v4bf*)&As[r][ac0] = p;            // one ds_store_b64
        }
        v8bf lo, hi;
        #pragma unroll
        for (int kk = 0; kk < 8; ++kk) {
            lo[kk] = (__bf16)bReg[kk];
            hi[kk] = (__bf16)bReg[8 + kk];
        }
        *(v8bf*)&Bs[bn][bk0 + 0] = lo;          // two ds_store_b128
        *(v8bf*)&Bs[bn][bk0 + 8] = hi;
    };

    globalLoad(0);
    ldsStore();
    __syncthreads();

    for (int kBase = 0; kBase < DM; kBase += BK) {
        const bool hasNext = (kBase + BK) < DM;
        if (hasNext) globalLoad(kBase + BK);   // overlap HBM latency with WMMAs

        // ---- A fragment: 16x32 bf16; two 16B LDS chunks per fragment ----
        Frag a;
        const int arow = wave * 16 + l16;
        a.q[0] = *(const v4ui*)&As[arow][half * 8];        // kk = half*8 + {0,2,4,6}
        a.q[1] = *(const v4ui*)&As[arow][16 + half * 8];   // kk = 16 + half*8 + {0,2,4,6}
        #pragma unroll
        for (int nt = 0; nt < 8; ++nt) {
            Frag b;
            const int bcol = nt * 16 + l16;
            b.q[0] = *(const v4ui*)&Bs[bcol][half * 8];
            b.q[1] = *(const v4ui*)&Bs[bcol][16 + half * 8];
            acc[nt] = __builtin_amdgcn_wmma_f32_16x16x32_bf16(
                false, a.v, false, b.v, (short)0, acc[nt], false, false);
        }
        __syncthreads();
        if (hasNext) {
            ldsStore();
            __syncthreads();
        }
    }

    // ---- epilogue: bias + activation, C/D layout: M = r + 8*half, N = lane%16 ----
    #pragma unroll
    for (int nt = 0; nt < 8; ++nt) {
        const int col = nBase + nt * 16 + l16;
        const float bv = bias[col];
        #pragma unroll
        for (int r = 0; r < 8; ++r) {
            int row = mBase + wave * 16 + r + half * 8;
            if (row < M) {
                float x = acc[nt][r] + bv;
                if (ACT == 1)      x = fmaxf(x, 0.f);
                else if (ACT == 2) x = gelu_exact(x);
                C[(size_t)row * DM + col] = x;
            }
        }
    }
}

// wave32 butterfly + one cross-wave LDS step; returns {mean, rsqrt(var+eps)}
static __device__ __forceinline__ float2 block_meanvar(float s, float ss) {
    #pragma unroll
    for (int off = 16; off > 0; off >>= 1) {
        s  += __shfl_xor(s, off, 32);
        ss += __shfl_xor(ss, off, 32);
    }
    __shared__ float2 part[8];
    const int lane = threadIdx.x & 31;
    const int wave = threadIdx.x >> 5;
    if (lane == 0) part[wave] = make_float2(s, ss);
    __syncthreads();
    float S = 0.f, SS = 0.f;
    #pragma unroll
    for (int w = 0; w < 8; ++w) { S += part[w].x; SS += part[w].y; }
    float mean = S * (1.f / DM);
    float var  = SS * (1.f / DM) - mean * mean;
    return make_float2(mean, rsqrtf(var + 1e-5f));
}

// out[row] = LN( sum_f emb[f, idx[row,f], :] )  (one block per row, 3 elems/thread)
__global__ __launch_bounds__(256) void embed_ln(
    const int* __restrict__ idx, int F, int V,
    const float* __restrict__ emb,
    const float* __restrict__ g, const float* __restrict__ b,
    float* __restrict__ out)
{
    const int row = blockIdx.x;
    const int t = threadIdx.x;
    float vals[3] = {0.f, 0.f, 0.f};
    for (int f = 0; f < F; ++f) {
        int id = idx[row * F + f];
        const float* e = emb + ((size_t)f * V + id) * DM;
        #pragma unroll
        for (int j = 0; j < 3; ++j) vals[j] += e[t + j * 256];
    }
    float s = 0.f, ss = 0.f;
    #pragma unroll
    for (int j = 0; j < 3; ++j) { s += vals[j]; ss += vals[j] * vals[j]; }
    float2 mv = block_meanvar(s, ss);
    #pragma unroll
    for (int j = 0; j < 3; ++j) {
        int d = t + j * 256;
        out[(size_t)row * DM + d] = (vals[j] - mv.x) * mv.y * g[d] + b[d];
    }
}

// out[row] = LN( z[row] + hprev[row] ) — in-place safe on hprev (reads before writes)
__global__ __launch_bounds__(256) void residual_ln(
    const float* __restrict__ z, const float* __restrict__ hprev,
    const float* __restrict__ g, const float* __restrict__ b,
    float* __restrict__ out)
{
    const int row = blockIdx.x;
    const int t = threadIdx.x;
    float vals[3];
    #pragma unroll
    for (int j = 0; j < 3; ++j) {
        int d = t + j * 256;
        vals[j] = z[(size_t)row * DM + d] + hprev[(size_t)row * DM + d];
    }
    float s = 0.f, ss = 0.f;
    #pragma unroll
    for (int j = 0; j < 3; ++j) { s += vals[j]; ss += vals[j] * vals[j]; }
    float2 mv = block_meanvar(s, ss);
    #pragma unroll
    for (int j = 0; j < 3; ++j) {
        int d = t + j * 256;
        out[(size_t)row * DM + d] = (vals[j] - mv.x) * mv.y * g[d] + b[d];
    }
}

// aggr[dst[e]] += relu(h[src[e]] + efeat[e]); one block (192 thr) per edge
__global__ __launch_bounds__(192) void edge_msg(
    const float* __restrict__ h, const float* __restrict__ efeat,
    const int* __restrict__ src, const int* __restrict__ dst,
    float* __restrict__ aggr, int E)
{
    const int e = blockIdx.x;
    if (e >= E) return;
    const int s = src[e], d = dst[e];
    const int c = threadIdx.x * 4;
    const float4 hv = *(const float4*)(h + (size_t)s * DM + c);
    const float4 ev = *(const float4*)(efeat + (size_t)e * DM + c);
    float4 m;
    m.x = fmaxf(hv.x + ev.x, 0.f);
    m.y = fmaxf(hv.y + ev.y, 0.f);
    m.z = fmaxf(hv.z + ev.z, 0.f);
    m.w = fmaxf(hv.w + ev.w, 0.f);
    float* a = aggr + (size_t)d * DM + c;
    atomicAdd(a + 0, m.x);
    atomicAdd(a + 1, m.y);
    atomicAdd(a + 2, m.z);
    atomicAdd(a + 3, m.w);
}

extern "C" void kernel_launch(void* const* d_in, const int* in_sizes, int n_in,
                              void* d_out, int out_size, void* d_ws, size_t ws_size,
                              hipStream_t stream) {
    const int*   x          = (const int*)  d_in[0];
    const int*   edge_attr  = (const int*)  d_in[1];
    const int*   edge_index = (const int*)  d_in[2];
    const float* atom_emb   = (const float*)d_in[3];
    const float* atom_ln_g  = (const float*)d_in[4];
    const float* atom_ln_b  = (const float*)d_in[5];
    const float* atom_w1    = (const float*)d_in[6];
    const float* atom_b1    = (const float*)d_in[7];
    const float* atom_w2    = (const float*)d_in[8];
    const float* atom_b2    = (const float*)d_in[9];
    const float* bond_emb   = (const float*)d_in[10];
    const float* bond_ln_g  = (const float*)d_in[11];
    const float* bond_ln_b  = (const float*)d_in[12];
    const float* bond_w1    = (const float*)d_in[13];
    const float* bond_b1    = (const float*)d_in[14];
    const float* bond_w2    = (const float*)d_in[15];
    const float* bond_b2    = (const float*)d_in[16];
    const float* conv_w1    = (const float*)d_in[17];
    const float* conv_b1    = (const float*)d_in[18];
    const float* conv_w2    = (const float*)d_in[19];
    const float* conv_b2    = (const float*)d_in[20];
    const float* ln_g       = (const float*)d_in[21];
    const float* ln_b       = (const float*)d_in[22];

    const int N = in_sizes[0] / 9;
    const int E = in_sizes[1] / 3;
    const int L = 4;

    float* ws    = (float*)d_ws;
    float* h     = ws;                           // N*DM
    float* aggr  = h     + (size_t)N * DM;       // N*DM
    float* efeat = aggr  + (size_t)N * DM;       // E*DM
    float* t0    = efeat + (size_t)E * DM;       // E*DM (scratch, max rows)
    float* t1    = t0    + (size_t)E * DM;       // E*DM (MLP hidden)

    const int* src = edge_index;
    const int* dst = edge_index + E;

    dim3 gN((unsigned)((N + BM - 1) / BM), DM / BN);
    dim3 gE((unsigned)((E + BM - 1) / BM), DM / BN);

    // AtomEncoder: embed-sum + LN -> Linear/GELU -> Linear -> h
    embed_ln<<<N, 256, 0, stream>>>(x, 9, 64, atom_emb, atom_ln_g, atom_ln_b, t0);
    gemm_wmma<2, false><<<gN, 256, 0, stream>>>(t0, nullptr, atom_w1, atom_b1, t1, N);
    gemm_wmma<0, false><<<gN, 256, 0, stream>>>(t1, nullptr, atom_w2, atom_b2, h, N);

    // BondEncoder -> efeat
    embed_ln<<<E, 256, 0, stream>>>(edge_attr, 3, 8, bond_emb, bond_ln_g, bond_ln_b, t0);
    gemm_wmma<2, false><<<gE, 256, 0, stream>>>(t0, nullptr, bond_w1, bond_b1, t1, E);
    gemm_wmma<0, false><<<gE, 256, 0, stream>>>(t1, nullptr, bond_w2, bond_b2, efeat, E);

    // GINE layers
    for (int l = 0; l < L; ++l) {
        hipMemsetAsync(aggr, 0, (size_t)N * DM * sizeof(float), stream);
        edge_msg<<<E, 192, 0, stream>>>(h, efeat, src, dst, aggr, E);
        // relu((h + aggr) @ W1 + b1): z = h + aggr fused into A-load
        gemm_wmma<1, true><<<gN, 256, 0, stream>>>(
            h, aggr, conv_w1 + (size_t)l * DM * DM, conv_b1 + (size_t)l * DM, t1, N);
        // gelu(hidden @ W2 + b2)
        gemm_wmma<2, false><<<gN, 256, 0, stream>>>(
            t1, nullptr, conv_w2 + (size_t)l * DM * DM, conv_b2 + (size_t)l * DM, t0, N);
        // h = LN(gelu_out + h)
        float* outp = (l == L - 1) ? (float*)d_out : h;
        residual_ln<<<N, 256, 0, stream>>>(t0, h, ln_g + (size_t)l * DM, ln_b + (size_t)l * DM, outp);
    }
}
